// MoEGate_51582557225385
// MI455X (gfx1250) — compile-verified, matrix operand order checked by reading, not run on hardware
//
#include <hip/hip_runtime.h>
#include <hip/hip_bf16.h>

typedef __attribute__((ext_vector_type(16))) __bf16 v16bf;
typedef __attribute__((ext_vector_type(8)))  __bf16 v8bf;
typedef __attribute__((ext_vector_type(8)))  float  v8f;

union BFrag { v16bf v; v8bf h[2]; };

constexpr int   kB        = 4;
constexpr int   kS        = 4096;
constexpr int   kH        = 4096;
constexpr int   kT        = kB * kS;        // 16384 tokens
constexpr int   kE        = 160;
constexpr int   kG        = 8;
constexpr int   kEperG    = kE / kG;        // 20
constexpr int   kTopkGrp  = 3;
constexpr int   kTopK     = 6;
constexpr float kScale    = 16.0f;
constexpr float kAlpha    = 0.001f;
constexpr float kM        = 3.0f;

// --------------------------------------------------------------------------
// Kernel 0b: one-shot W fp32 -> bf16 (1.3 MB, L2-resident thereafter).
// --------------------------------------------------------------------------
__global__ __launch_bounds__(256) void k_convert_w(
    const float* __restrict__ W, __bf16* __restrict__ Wbf)
{
  const int i = (blockIdx.x * 256 + threadIdx.x) * 8;
  if (i >= kE * kH) return;
  v8bf o;
#pragma unroll
  for (int j = 0; j < 8; ++j) o[j] = (__bf16)W[i + j];
  *(v8bf*)(Wbf + i) = o;
}

// --------------------------------------------------------------------------
// Kernel 1: logits[T,160] = X[T,4096] @ W[160,4096]^T   (bf16 WMMA, f32 acc)
// 256 threads = 8 waves. Each wave owns a 16-row slab of X and all 10 N
// tiles. The 160x32 bf16 W tile per K step is staged once per block into
// double-buffered LDS; B fragments are additionally double-buffered in
// registers so ds_load of tile t+1 overlaps the WMMA of tile t (DS returns
// in order, so the wait degrades from dscnt==0 to dscnt<=2). A fragments
// are converted fp32->bf16 in registers; X streams from HBM exactly once,
// which is this GEMM's roofline floor (~268 MB @ 23.3 TB/s).
// --------------------------------------------------------------------------
__global__ __launch_bounds__(256) void k_gemm_logits(
    const float* __restrict__ X, const __bf16* __restrict__ Wbf,
    float* __restrict__ logits)
{
  __shared__ alignas(16) __bf16 sw[2][kE * 32];   // 2 x 10 KB

  const int tid  = threadIdx.x;
  const int wave = tid >> 5;                // 0..7
  const int lane = tid & 31;
  const int row0 = blockIdx.x * 128 + wave * 16;
  const int m    = lane & 15;               // M (A) / N (B) within tile
  const int ko   = (lane >> 4) * 8;         // K sub-offset from lane half

  const float* xrow = X + (size_t)(row0 + m) * kH + ko;

  // Stage tile kb = 0: 160 rows x 32 K of bf16 = 640 chunks of 8 bf16.
  for (int c = tid; c < kE * 4; c += 256) {
    const int row = c >> 2, part = c & 3;
    *(v8bf*)&sw[0][row * 32 + part * 8] =
        *(const v8bf*)(Wbf + (size_t)row * kH + part * 8);
  }

  v8f acc[10] = {};                         // 10 x 16x16 f32 accumulators
  int p = 0;
  for (int kb = 0; kb < kH; kb += 32, p ^= 1) {
    __syncthreads();
    // Stage the next W tile into the other LDS buffer.
    if (kb + 32 < kH) {
      const int kn = kb + 32;
      for (int c = tid; c < kE * 4; c += 256) {
        const int row = c >> 2, part = c & 3;
        *(v8bf*)&sw[p ^ 1][row * 32 + part * 8] =
            *(const v8bf*)(Wbf + (size_t)row * kH + kn + part * 8);
      }
    }

    const float* ap = xrow + kb;
    __builtin_prefetch(ap + 96, 0, 0);      // global_prefetch_b8 on X stream

    // A fragment: 16-bit A 16x32 layout (lane half selects K 0..7 / 8..15,
    // upper VGPRs hold K+16).
    v16bf a;
#pragma unroll
    for (int j = 0; j < 8; ++j) {
      a[j]     = (__bf16)ap[j];
      a[j + 8] = (__bf16)ap[16 + j];
    }

    // Register-double-buffered B fragments over the 10 N tiles.
    const __bf16* swp = &sw[p][m * 32 + ko];
    BFrag bf[2];
    bf[0].h[0] = *(const v8bf*)swp;         // ds_load_b128
    bf[0].h[1] = *(const v8bf*)(swp + 16);  // ds_load_b128
#pragma unroll
    for (int t = 0; t < 10; ++t) {
      const int cur = t & 1;
      if (t < 9) {
        const __bf16* spn = swp + (t + 1) * 16 * 32;
        bf[cur ^ 1].h[0] = *(const v8bf*)spn;
        bf[cur ^ 1].h[1] = *(const v8bf*)(spn + 16);
      }
      acc[t] = __builtin_amdgcn_wmma_f32_16x16x32_bf16(
          /*neg_a=*/false, a, /*neg_b=*/false, bf[cur].v,
          /*c_mod=*/(short)0, acc[t], /*reuse_a=*/false, /*reuse_b=*/false);
    }
  }

  // f32 C/D layout: VGPR r -> M = r + (lane>=16 ? 8 : 0), N = lane & 15.
  const int n_lane = lane & 15;
  const int m_off  = (lane >> 4) * 8;
#pragma unroll
  for (int t = 0; t < 10; ++t)
#pragma unroll
    for (int r = 0; r < 8; ++r)
      logits[(size_t)(row0 + m_off + r) * kE + t * 16 + n_lane] = acc[t][r];
}

// --------------------------------------------------------------------------
// Kernel 2: per-token softmax + group-limited top-k. One thread per token;
// the 160 logits live in L2. Overwrites logits with softmax scores (needed
// by the mean-score reduction), emits idx/weight, accumulates counts[b][e].
// --------------------------------------------------------------------------
__global__ __launch_bounds__(256) void k_softmax_topk(
    float* __restrict__ scores, float* __restrict__ out_idx,
    float* __restrict__ out_w, float* __restrict__ counts)
{
  const int t = blockIdx.x * blockDim.x + threadIdx.x;
  if (t >= kT) return;
  float* lg = scores + (size_t)t * kE;

  float mx = -3.4e38f;
  for (int e = 0; e < kE; ++e) mx = fmaxf(mx, lg[e]);
  float sum = 0.f;
  for (int e = 0; e < kE; ++e) sum += __expf(lg[e] - mx);
  const float inv = 1.f / sum;

  float gmax[kG];
#pragma unroll
  for (int g = 0; g < kG; ++g) gmax[g] = -1.f;
  for (int e = 0; e < kE; ++e) {
    const float sc = __expf(lg[e] - mx) * inv;
    lg[e] = sc;
    const int g = e / kEperG;
    gmax[g] = fmaxf(gmax[g], sc);
  }

  unsigned gsel = 0;
  for (int r = 0; r < kTopkGrp; ++r) {
    float best = -1.f; int bi = 0;
#pragma unroll
    for (int g = 0; g < kG; ++g)
      if (!((gsel >> g) & 1u) && gmax[g] > best) { best = gmax[g]; bi = g; }
    gsel |= 1u << bi;
  }

  int sel[kTopK];
  const int b = t / kS;
  for (int r = 0; r < kTopK; ++r) {
    float best = -1.f; int bi = 0;
    for (int e = 0; e < kE; ++e) {
      if (!((gsel >> (e / kEperG)) & 1u)) continue;
      bool taken = false;
#pragma unroll
      for (int q = 0; q < kTopK; ++q) taken |= (q < r) && (sel[q] == e);
      if (taken) continue;
      const float sc = lg[e];
      if (sc > best) { best = sc; bi = e; }
    }
    sel[r] = bi;
    out_idx[(size_t)t * kTopK + r] = (float)bi;
    out_w[(size_t)t * kTopK + r]   = best * kScale;
    atomicAdd(&counts[b * kE + bi], 1.0f);
  }
}

// --------------------------------------------------------------------------
// Kernel 3: msum[b][e] = sum over s of scores[b,s,e]. One block per (b,e).
// --------------------------------------------------------------------------
__global__ __launch_bounds__(256) void k_mean_scores(
    const float* __restrict__ scores, float* __restrict__ msum)
{
  const int b = blockIdx.x / kE;
  const int e = blockIdx.x % kE;
  __shared__ float red[256];
  float acc = 0.f;
  for (int s = threadIdx.x; s < kS; s += 256)
    acc += scores[((size_t)b * kS + s) * kE + e];
  red[threadIdx.x] = acc;
  __syncthreads();
  for (int off = 128; off > 0; off >>= 1) {
    if ((int)threadIdx.x < off) red[threadIdx.x] += red[threadIdx.x + off];
    __syncthreads();
  }
  if (threadIdx.x == 0) msum[b * kE + e] = red[0];
}

// --------------------------------------------------------------------------
// Kernel 4: scalar aux-loss epilogue over 2 x [4,160] buffers.
// --------------------------------------------------------------------------
__global__ void k_aux(const float* __restrict__ counts,
                      const float* __restrict__ msum,
                      float* __restrict__ aux_out)
{
  if (threadIdx.x != 0 || blockIdx.x != 0) return;
  float aux_e = 0.f, aux_d = 0.f, aux_c = 0.f;
  const float ce_norm  = (float)kE / ((float)kS * (float)kTopK); // 1/(s*k/E)
  const float ce2_norm = (float)kG / (kM * (float)kS);           // 1/(M*s/G)
  for (int b = 0; b < kB; ++b) {
    float se = 0.f;
    float ce_g[kG] = {}, ms_g[kG] = {}, cnt_g[kG] = {};
    for (int e = 0; e < kE; ++e) {
      const float cnt = counts[b * kE + e];
      const float ce  = cnt * ce_norm;
      const float ms  = msum[b * kE + e] / (float)kS;
      se += ce * ms;
      const int g = e / kEperG;
      ce_g[g] += ce / (float)kEperG;
      ms_g[g] += ms / (float)kEperG;
      cnt_g[g] += cnt;
    }
    aux_e += se;
    for (int g = 0; g < kG; ++g) {
      aux_d += ce_g[g] * ms_g[g];
      aux_c += (cnt_g[g] * ce2_norm) * ms_g[g];
    }
  }
  const float invB = 1.f / (float)kB;
  *aux_out = (aux_e * invB + aux_d * invB + aux_c * invB) * kAlpha;
}

__global__ void k_zero(float* __restrict__ p, int n)
{
  const int i = blockIdx.x * blockDim.x + threadIdx.x;
  if (i < n) p[i] = 0.f;
}

extern "C" void kernel_launch(void* const* d_in, const int* in_sizes, int n_in,
                              void* d_out, int out_size, void* d_ws, size_t ws_size,
                              hipStream_t stream) {
  const float* X = (const float*)d_in[0];   // [B,S,H] fp32
  const float* W = (const float*)d_in[1];   // [E,H]   fp32
  float* out     = (float*)d_out;           // idx[T*6] | weight[T*6] | aux[1]

  float*  logits = (float*)d_ws;                    // kT*kE f32 (then scores)
  float*  counts = logits + (size_t)kT * kE;        // kB*kE
  float*  msum   = counts + kB * kE;                // kB*kE
  __bf16* wbf    = (__bf16*)(msum + kB * kE);       // kE*kH bf16 (1.3 MB)

  (void)in_sizes; (void)n_in; (void)out_size; (void)ws_size;

  k_zero<<<(2 * kB * kE + 255) / 256, 256, 0, stream>>>(counts, 2 * kB * kE);
  k_convert_w<<<(kE * kH / 8 + 255) / 256, 256, 0, stream>>>(W, wbf);
  k_gemm_logits<<<kT / 128, 256, 0, stream>>>(X, wbf, logits);
  k_softmax_topk<<<kT / 256, 256, 0, stream>>>(
      logits, out, out + (size_t)kT * kTopK, counts);
  k_mean_scores<<<kB * kE, 256, 0, stream>>>(logits, msum);
  k_aux<<<1, 32, 0, stream>>>(counts, msum, out + 2 * (size_t)kT * kTopK);
}
